// SwitchSparseMoe_8400956031009
// MI455X (gfx1250) — matrix operand order repaired and checked by reading, not compiled, not run on hardware
//
#include <hip/hip_runtime.h>
#include <hip/hip_bf16.h>

// ---------------- problem constants ----------------
#define BB 8
#define SS 2048
#define DD 1024
#define EE 8
#define FF 2048
#define CC 320          // expert capacity
#define RMS_EPS 1e-6f

typedef __attribute__((ext_vector_type(16))) _Float16 v16h;
typedef __attribute__((ext_vector_type(8)))  float    v8f;

// ---------------- 1. weight convert f32 -> f16 with transpose ----------------
// src: [E][K][N] f32 row-major  ->  dst: [E][N][K] f16 (i.e. B^T, K-contiguous)
__global__ __launch_bounds__(256)
void cvt_transpose_kernel(const float* __restrict__ src,
                          _Float16* __restrict__ dst,
                          const int K, const int N)
{
    __shared__ float tile[32][33];
    const int e  = blockIdx.z;
    const int k0 = blockIdx.y * 32;
    const int n0 = blockIdx.x * 32;
    const int tr = threadIdx.x >> 3;
    const int tc = (threadIdx.x & 7) * 4;

    const float4 v = *(const float4*)(src + ((long)e * K + k0 + tr) * N + n0 + tc);
    tile[tr][tc + 0] = v.x;
    tile[tr][tc + 1] = v.y;
    tile[tr][tc + 2] = v.z;
    tile[tr][tc + 3] = v.w;
    __syncthreads();

    union { _Float16 h[4]; uint2 u; } t;
#pragma unroll
    for (int j = 0; j < 4; ++j) t.h[j] = (_Float16)tile[tc + j][tr];
    *(uint2*)(dst + ((long)e * N + n0 + tr) * K + k0 + tc) = t.u;
}

// ---------------- 2. RMSNorm + gate + softmax/argmax ----------------
__global__ __launch_bounds__(256)
void rmsnorm_gate_kernel(const float* __restrict__ data,
                         const float* __restrict__ gate_w,   // [D,E] row-major
                         const float* __restrict__ rms_w,
                         float* __restrict__ norm,           // [B*S, D]
                         float* __restrict__ logits_out,     // [B*S, E]
                         float* __restrict__ maxprob,        // [B*S]
                         int*   __restrict__ eidx)           // [B*S]
{
    __shared__ float red[256];
    __shared__ float lsh[EE];
    const long tok = blockIdx.x;
    const int  tid = threadIdx.x;
    const long base = tok * DD + tid * 4;

    float4 x = *(const float4*)(data + base);
    float ss = x.x * x.x + x.y * x.y + x.z * x.z + x.w * x.w;

    red[tid] = ss; __syncthreads();
    for (int off = 128; off > 0; off >>= 1) {
        if (tid < off) red[tid] += red[tid + off];
        __syncthreads();
    }
    const float rstd = rsqrtf(red[0] * (1.0f / (float)DD) + RMS_EPS);
    __syncthreads();

    const float4 w = *(const float4*)(rms_w + tid * 4);
    float n0 = x.x * rstd * w.x, n1 = x.y * rstd * w.y;
    float n2 = x.z * rstd * w.z, n3 = x.w * rstd * w.w;
    float4 nv = {n0, n1, n2, n3};
    *(float4*)(norm + base) = nv;

    float p[EE];
#pragma unroll
    for (int e = 0; e < EE; ++e) p[e] = 0.f;
    const float nj[4] = {n0, n1, n2, n3};
#pragma unroll
    for (int j = 0; j < 4; ++j) {
        const int d = tid * 4 + j;
        const float4 g0 = *(const float4*)(gate_w + (long)d * EE);
        const float4 g1 = *(const float4*)(gate_w + (long)d * EE + 4);
        p[0] += nj[j] * g0.x; p[1] += nj[j] * g0.y;
        p[2] += nj[j] * g0.z; p[3] += nj[j] * g0.w;
        p[4] += nj[j] * g1.x; p[5] += nj[j] * g1.y;
        p[6] += nj[j] * g1.z; p[7] += nj[j] * g1.w;
    }
    for (int e = 0; e < EE; ++e) {
        __syncthreads();
        red[tid] = p[e]; __syncthreads();
        for (int off = 128; off > 0; off >>= 1) {
            if (tid < off) red[tid] += red[tid + off];
            __syncthreads();
        }
        if (tid == 0) lsh[e] = red[0];
    }
    __syncthreads();
    if (tid == 0) {
        float m = lsh[0]; int am = 0;
        for (int e = 1; e < EE; ++e) { if (lsh[e] > m) { m = lsh[e]; am = e; } }
        float z = 0.f;
        for (int e = 0; e < EE; ++e) z += __expf(lsh[e] - m);
        for (int e = 0; e < EE; ++e) logits_out[tok * EE + e] = lsh[e];
        maxprob[tok] = 1.0f / z;
        eidx[tok]    = am;
    }
}

// ---------------- 3. routing scan (capacity cumsum) ----------------
__global__ void route_kernel(const int* __restrict__ eidx,
                             int*   __restrict__ pos_sel,
                             int*   __restrict__ dispatch_idx,   // [B,E,C]
                             float* __restrict__ expert_index_out)
{
    const int b = blockIdx.x;
    const int e = threadIdx.x;
    int* dslot = dispatch_idx + ((long)b * EE + e) * CC;
    for (int c = 0; c < CC; ++c) dslot[c] = -1;
    int cnt = 0;
    const long tb = (long)b * SS;
    for (int s = 0; s < SS; ++s) {
        if (eidx[tb + s] == e) {
            if (cnt < CC) {
                dslot[cnt] = s;
                pos_sel[tb + s] = cnt;
                expert_index_out[tb + s] = (float)e;
                ++cnt;
            } else {
                pos_sel[tb + s] = -1;
                expert_index_out[tb + s] = 0.0f;
            }
        }
    }
}

// ---------------- 4. dispatch gather norm -> f16 buffer ----------------
__global__ __launch_bounds__(256)
void dispatch_kernel(const float* __restrict__ norm,
                     const int*   __restrict__ dispatch_idx,
                     _Float16*    __restrict__ Xd)       // [B,E,C,D]
{
    const long row = blockIdx.x;
    const int  tid = threadIdx.x;
    const int  b   = (int)(row / (EE * CC));
    const int  s   = dispatch_idx[row];
    union { _Float16 h[4]; uint2 u; } t;
    if (s >= 0) {
        const float4 v = *(const float4*)(norm + ((long)b * SS + s) * DD + tid * 4);
        t.h[0] = (_Float16)v.x; t.h[1] = (_Float16)v.y;
        t.h[2] = (_Float16)v.z; t.h[3] = (_Float16)v.w;
    } else {
        t.u = make_uint2(0u, 0u);
    }
    *(uint2*)(Xd + row * DD + tid * 4) = t.u;
}

// ---------------- 5/6. tiled WMMA GEMM ----------------
// per z=(b*E+e):  Out[320,N] = A[320,K] x B[K,N], B given transposed [N][K]
// block tile 64x128x32, 8 waves (2x4), wave tile 32x32 = 4 WMMAs/K-step
#define BM 64
#define BN 128
#define BK 32
#define LDA 48   // halves per LDS row: 32 + 16 pad -> 96 B, keeps 32B alignment
#define LDB 48

__global__ __launch_bounds__(256)
void moe_gemm_kernel(const _Float16* __restrict__ A,     // [z][320][K]
                     const _Float16* __restrict__ Bt,    // [e][N][K]
                     _Float16* __restrict__ OutH,
                     float*    __restrict__ OutF,
                     const int K, const int N, const int applyGelu)
{
    __shared__ _Float16 As[BM * LDA];   // 6 KB
    __shared__ _Float16 Bs[BN * LDB];   // 12 KB

    const int z = blockIdx.z;
    const int e = z & (EE - 1);
    const _Float16* Ab = A  + (long)z * CC * K;
    const _Float16* Bb = Bt + (long)e * N * K;

    const int m0b = blockIdx.y * BM;
    const int n0b = blockIdx.x * BN;

    const int tid  = threadIdx.x;
    const int wid  = tid >> 5;
    const int lane = tid & 31;
    const int half = lane >> 4;
    const int r    = lane & 15;
    const int wm   = (wid >> 2) * 32;   // 0 or 32   (wave: 32M x 32N)
    const int wn   = (wid & 3)  * 32;   // 0,32,64,96

    // A loader: 64 rows x 32 halves, 8 halves/thread (one uint4)
    // row layout in LDS: K-groups contiguous [K0-7 | K8-15 | K16-23 | K24-31]
    const int ar = tid >> 2;
    const int g  = tid & 3;                       // K-group of this chunk
    const int ng = g;                             // identity
    // B loader: 128 rows x 32 halves (K-contig), 16 halves/thread
    const int brow = tid >> 1;
    const int bcol = (tid & 1) * 16;

    const _Float16* agp = Ab + (long)(m0b + ar) * K + g * 8;
    const _Float16* bgp = Bb + (long)(n0b + brow) * K + bcol;
    _Float16* asp = &As[ar * LDA + ng * 8];
    _Float16* bsp = &Bs[brow * LDB + bcol];

    v8f acc00 = {}, acc01 = {}, acc10 = {}, acc11 = {};

    for (int k0 = 0; k0 < K; k0 += BK) {
        __syncthreads();
        *(uint4*)asp       = *(const uint4*)(agp + k0);
        *(uint4*)bsp       = *(const uint4*)(bgp + k0);
        *(uint4*)(bsp + 8) = *(const uint4*)(bgp + k0 + 8);
        if (k0 + BK < K) {   // -> global_prefetch_b8
            __builtin_prefetch(agp + k0 + BK, 0, 1);
            __builtin_prefetch(bgp + k0 + BK, 0, 1);
        }
        __syncthreads();

        // fragments: contiguous 32B per lane, 32B aligned
        const v16h a0 = *(const v16h*)&As[(wm + r) * LDA + half * 16];
        const v16h a1 = *(const v16h*)&As[(wm + 16 + r) * LDA + half * 16];
        const v16h b0 = *(const v16h*)&Bs[(wn + r) * LDB + half * 16];
        const v16h b1 = *(const v16h*)&Bs[(wn + 16 + r) * LDB + half * 16];

        acc00 = __builtin_amdgcn_wmma_f32_16x16x32_f16(
            false, a0, false, b0, (short)0, acc00, false, false);
        acc01 = __builtin_amdgcn_wmma_f32_16x16x32_f16(
            false, a0, false, b1, (short)0, acc01, false, false);
        acc10 = __builtin_amdgcn_wmma_f32_16x16x32_f16(
            false, a1, false, b0, (short)0, acc10, false, false);
        acc11 = __builtin_amdgcn_wmma_f32_16x16x32_f16(
            false, a1, false, b1, (short)0, acc11, false, false);
    }

    // C/D layout: vgpr rr -> M = rr + 8*half; lane -> N = r
    const long obase = (long)z * CC * N;
    const int  nc0   = n0b + wn + r;
    if (applyGelu) {
#pragma unroll
        for (int rr = 0; rr < 8; ++rr) {
            const int mr0 = m0b + wm + rr + 8 * half;
            const int mr1 = mr0 + 16;
            const float xs[4] = {acc00[rr], acc01[rr], acc10[rr], acc11[rr]};
            _Float16 gs[4];
#pragma unroll
            for (int q = 0; q < 4; ++q) {
                const float x = xs[q];
                const float u = 0.7978845608028654f * (x + 0.044715f * x * x * x);
                const float t = 1.0f - 2.0f / (1.0f + __expf(2.0f * u)); // tanh(u)
                gs[q] = (_Float16)(0.5f * x * (1.0f + t));
            }
            OutH[obase + (long)mr0 * N + nc0]      = gs[0];
            OutH[obase + (long)mr0 * N + nc0 + 16] = gs[1];
            OutH[obase + (long)mr1 * N + nc0]      = gs[2];
            OutH[obase + (long)mr1 * N + nc0 + 16] = gs[3];
        }
    } else {
#pragma unroll
        for (int rr = 0; rr < 8; ++rr) {
            const int mr0 = m0b + wm + rr + 8 * half;
            const int mr1 = mr0 + 16;
            OutF[obase + (long)mr0 * N + nc0]      = acc00[rr];
            OutF[obase + (long)mr0 * N + nc0 + 16] = acc01[rr];
            OutF[obase + (long)mr1 * N + nc0]      = acc10[rr];
            OutF[obase + (long)mr1 * N + nc0 + 16] = acc11[rr];
        }
    }
}

// ---------------- 7. combine ----------------
__global__ __launch_bounds__(256)
void combine_kernel(const float* __restrict__ data,
                    const float* __restrict__ norm,
                    const float* __restrict__ outbuf,   // [B,E,C,D]
                    const int*   __restrict__ eidx,
                    const int*   __restrict__ pos_sel,
                    const float* __restrict__ maxprob,
                    float* __restrict__ out_final)
{
    const long tok = blockIdx.x;
    const int  tid = threadIdx.x;
    const int  b   = (int)(tok / SS);
    const int  e   = eidx[tok];
    const int  p   = pos_sel[tok];
    const float mp = maxprob[tok];

    const float* src = (p >= 0)
        ? outbuf + (((long)(b * EE + e) * CC + p) * DD)
        : norm + tok * DD;

    const long d = tid * 4;
    float4 x = *(const float4*)(data + tok * DD + d);
    float4 v = *(const float4*)(src + d);
    float4 o = {x.x + mp * v.x, x.y + mp * v.y, x.z + mp * v.z, x.w + mp * v.w};
    *(float4*)(out_final + tok * DD + d) = o;
}

// ---------------- host launcher ----------------
extern "C" void kernel_launch(void* const* d_in, const int* in_sizes, int n_in,
                              void* d_out, int out_size, void* d_ws, size_t ws_size,
                              hipStream_t stream) {
    const float* data   = (const float*)d_in[0];   // [B,S,D]
    const float* gate_w = (const float*)d_in[1];   // [D,E]
    const float* w1     = (const float*)d_in[2];   // [E,D,F]
    const float* w2     = (const float*)d_in[3];   // [E,F,D]
    const float* rms_w  = (const float*)d_in[4];   // [D]

    float* out_final  = (float*)d_out;
    float* out_logits = out_final + (size_t)BB * SS * DD;
    float* out_eidx   = out_logits + (size_t)BB * SS * EE;

    char* ws = (char*)d_ws;
    size_t off = 0;
    auto alloc = [&](size_t bytes) -> char* {
        char* p = ws + off;
        off = (off + bytes + 255) & ~(size_t)255;
        return p;
    };
    float*    norm    = (float*)   alloc((size_t)BB * SS * DD * 4);
    _Float16* w1t     = (_Float16*)alloc((size_t)EE * DD * FF * 2);  // [E][F][D]
    _Float16* w2t     = (_Float16*)alloc((size_t)EE * FF * DD * 2);  // [E][D][F]
    _Float16* Xd      = (_Float16*)alloc((size_t)BB * EE * CC * DD * 2);
    _Float16* H       = (_Float16*)alloc((size_t)BB * EE * CC * FF * 2);
    float*    outbuf  = (float*)   alloc((size_t)BB * EE * CC * DD * 4);
    float*    maxprob = (float*)   alloc((size_t)BB * SS * 4);
    int*      eidx    = (int*)     alloc((size_t)BB * SS * 4);
    int*      pos_sel = (int*)     alloc((size_t)BB * SS * 4);
    int*      didx    = (int*)     alloc((size_t)BB * EE * CC * 4);
    (void)ws_size; (void)in_sizes; (void)n_in; (void)out_size;

    // 1. weights -> f16, transposed to [N][K] per expert
    {
        dim3 g1(FF / 32, DD / 32, EE);   // w1: K=D, N=F -> w1t [E][F][D]
        cvt_transpose_kernel<<<g1, 256, 0, stream>>>(w1, w1t, DD, FF);
        dim3 g2(DD / 32, FF / 32, EE);   // w2: K=F, N=D -> w2t [E][D][F]
        cvt_transpose_kernel<<<g2, 256, 0, stream>>>(w2, w2t, FF, DD);
    }
    // 2. rmsnorm + gate
    rmsnorm_gate_kernel<<<BB * SS, 256, 0, stream>>>(
        data, gate_w, rms_w, norm, out_logits, maxprob, eidx);
    // 3. routing
    route_kernel<<<BB, EE, 0, stream>>>(eidx, pos_sel, didx, out_eidx);
    // 4. dispatch gather
    dispatch_kernel<<<BB * EE * CC, 256, 0, stream>>>(norm, didx, Xd);
    // 5. GEMM1 + gelu : [320,1024] x [1024,2048] -> H f16
    {
        dim3 grid(FF / BN, CC / BM, BB * EE);   // (16,5,64)
        moe_gemm_kernel<<<grid, 256, 0, stream>>>(Xd, w1t, H, nullptr, DD, FF, 1);
    }
    // 6. GEMM2 : [320,2048] x [2048,1024] -> outbuf f32
    {
        dim3 grid(DD / BN, CC / BM, BB * EE);   // (8,5,64)
        moe_gemm_kernel<<<grid, 256, 0, stream>>>(H, w2t, nullptr, outbuf, FF, DD, 0);
    }
    // 7. combine
    combine_kernel<<<BB * SS, 256, 0, stream>>>(
        data, norm, outbuf, eidx, pos_sel, maxprob, out_final);
}
